// _RFCNHead_56100862820588
// MI455X (gfx1250) — matrix operand order repaired and checked by reading, not compiled, not run on hardware
//
#include <hip/hip_runtime.h>

// R-FCN head for MI455X (gfx1250, wave32, WMMA + TDM double-buffered).
//
// Phase 1: sm[3969][10000] = conv_w[3969][1024] * x[1024][10000] + b
//   -> 81.3 GFLOP GEMM via v_wmma_f32_16x16x32_f16 (f16 in, f32 acc).
//   Both operands are [rows][1024] f16 with contiguous 64B tile rows; the
//   Tensor Data Mover stages tiles into LDS (pad_enable reproduces the 80B
//   padded LDS row stride) with ping-pong buffering: DMA of step i+1 overlaps
//   the WMMAs of step i (in-order TENSORcnt, wait <= 2).
// Phase 2: PSROIAlign pooling; sm (159MB fp32) is L2-resident (192MB L2).

#define IN_CH 1024
#define N_T   81
#define KGRID 7
#define HFM   100
#define WFM   100
#define NPIX  (HFM*WFM)                 // 10000
#define NPIX_PAD 10112                  // 79 * 128
#define R_REG 1024
#define M_OUT (N_T*KGRID*KGRID)         // 3969
#define M_PAD 4032                      // 63 * 64

#define BLK_M 64
#define BLK_N 128
#define BLK_K 32
#define LDS_STRIDE 40                   // 32 halves + 8 pad (80B, 16B aligned)
#define A_LDS_HALVES (BLK_M * LDS_STRIDE)   // 2560 halves = 5120 B
#define B_LDS_HALVES (BLK_N * LDS_STRIDE)   // 5120 halves = 10240 B
#define BUF_HALVES   (A_LDS_HALVES + B_LDS_HALVES)   // 7680 halves = 15360 B

typedef __attribute__((ext_vector_type(16))) _Float16 v16h;
typedef __attribute__((ext_vector_type(8)))  float    v8f;
typedef unsigned int v4u __attribute__((ext_vector_type(4)));
typedef int          v8i __attribute__((ext_vector_type(8)));
typedef int          v4i __attribute__((ext_vector_type(4)));

#if __has_builtin(__builtin_amdgcn_tensor_load_to_lds) && defined(__gfx1250__)
#define USE_TDM 1
#else
#define USE_TDM 0
#endif

// ---------------------------------------------------------------- converters
__global__ __launch_bounds__(256)
void rfcn_cvt_w(const float* __restrict__ w, _Float16* __restrict__ Ah, int n) {
    int i = blockIdx.x * blockDim.x + threadIdx.x;
    if (i >= n) return;                       // n = M_PAD*IN_CH
    int row = i / IN_CH;
    Ah[i] = (row < M_OUT) ? (_Float16)w[i] : (_Float16)0.f;  // zero-pad rows
}

// x (1024 ch, 10000 px) f32 -> BhT (10112 px, 1024 ch) f16, tiled transpose.
__global__ __launch_bounds__(256)
void rfcn_cvt_xT(const float* __restrict__ x, _Float16* __restrict__ BhT) {
    __shared__ _Float16 tile[32][33];
    const int p0 = blockIdx.x * 32;   // pixel base (0..10080)
    const int c0 = blockIdx.y * 32;   // channel base
    const int tx = threadIdx.x;       // 0..31
    const int ty = threadIdx.y;       // 0..7
#pragma unroll
    for (int j = 0; j < 4; ++j) {
        const int crel = ty + j * 8;
        const int p = p0 + tx;
        tile[crel][tx] = (p < NPIX) ? (_Float16)x[(size_t)(c0 + crel) * NPIX + p]
                                    : (_Float16)0.f;
    }
    __syncthreads();
#pragma unroll
    for (int j = 0; j < 4; ++j) {
        const int prel = ty + j * 8;
        BhT[(size_t)(p0 + prel) * IN_CH + c0 + tx] = tile[tx][prel];
    }
}

// ------------------------------------------------------- TDM descriptor pack
#if USE_TDM
// D# group0 (ISA 8.3): count=1, lds_addr, global_addr[56:0], type=2.
static __device__ __forceinline__ v4u tdm_group0(unsigned lds_addr,
                                                 unsigned long long ga) {
    v4u g;
    g.x = 1u;                                   // count=1, not gather/restore
    g.y = lds_addr;                             // LDS byte address
    g.z = (unsigned)ga;                         // global_addr[31:0]
    g.w = ((unsigned)(ga >> 32) & 0x01FFFFFFu)  // global_addr[56:32]
          | (2u << 30);                         // type=2 ("image")
    return g;
}
// D# group1 (ISA 8.4): data_size=2B, pad 4 dwords per 16 dwords (64B rows ->
// 80B LDS stride), 2D tile (tile_dim0 x tile_dim1), stride0 in elements.
static __device__ __forceinline__ v8i tdm_group1(unsigned tensor_d0, unsigned tensor_d1,
                                                 unsigned tile_d0,   unsigned tile_d1,
                                                 unsigned long long stride0) {
    v8i g;
    g[0] = (int)((1u << 16)        // data_size = 1 -> 2 bytes
               | (1u << 20)        // pad_enable
               | (3u << 22)        // pad_interval: 16 dwords (64B)
               | (3u << 25));      // pad_amount:   4 dwords (16B)
    g[1] = (int)((tensor_d0 & 0xFFFFu) << 16);                 // dim0[15:0]
    g[2] = (int)((tensor_d0 >> 16) | ((tensor_d1 & 0xFFFFu) << 16));
    g[3] = (int)((tensor_d1 >> 16) | (tile_d0 << 16));         // tile_dim0
    g[4] = (int)(tile_d1);                                     // tile_dim1, tile_dim2=0
    g[5] = (int)(unsigned)stride0;                             // dim0_stride[31:0]
    g[6] = (int)(unsigned)(stride0 >> 32);                     // dim0_stride[47:32]
    g[7] = 0;
    return g;
}
#endif

// ---------------------------------------------------------------- WMMA GEMM
// Block tile 64(M) x 128(N), K-step 32, 8 waves; wave (wm,wn) owns a 16x64
// strip = 4 f32 accumulators, one A fragment reused across 4 WMMAs.
// Fragment layouts per CDNA5 ISA 7.12.2 (A 16x32 K-interleave; B per-lane
// columns; C/D VGPR r -> M=r / r+8, N=lane&15).
__global__ __launch_bounds__(256)
void rfcn_gemm_wmma(const _Float16* __restrict__ Ah,   // [M_PAD][IN_CH]
                    const _Float16* __restrict__ BhT,  // [NPIX_PAD][IN_CH]
                    const float*    __restrict__ bias, // [M_OUT]
                    float*          __restrict__ sm)   // [M_OUT][NPIX]
{
    __shared__ _Float16 LDSBUF[2 * BUF_HALVES];        // ping-pong buffers

    const int tid    = threadIdx.x;
    const int lane   = tid & 31;
    const int wid    = tid >> 5;
    const int wave_m = wid & 3;
    const int wave_n = wid >> 2;

    const int m0 = blockIdx.y * BLK_M;
    const int n0 = blockIdx.x * BLK_N;

    v8f acc[4] = {};

    const int lrow  = wave_m * 16 + (lane & 15);
    const int a_klo = (lane < 16) ? 0 : 8;      // A K-interleave
    const int a_khi = a_klo + 16;
    const int b_klo = (lane < 16) ? 0 : 8;      // B K-interleave (same as A)

#if USE_TDM
    const v4i z4 = {0, 0, 0, 0};
    const v8i z8 = {0, 0, 0, 0, 0, 0, 0, 0};
    // Issue the A+B tile pair for K-step k into ping-pong buffer `ph`.
    auto issue_pair = [&](int k, int ph) {
        const unsigned ldsA = (unsigned)(ph * BUF_HALVES * 2);
        const unsigned ldsB = ldsA + (unsigned)(A_LDS_HALVES * 2);
        const unsigned long long gaA =
            (unsigned long long)Ah + ((size_t)m0 * IN_CH + k) * 2u;
        const unsigned long long gaB =
            (unsigned long long)BhT + ((size_t)n0 * IN_CH + k) * 2u;
        __builtin_amdgcn_tensor_load_to_lds(
            tdm_group0(ldsA, gaA),
            tdm_group1(IN_CH, M_PAD, BLK_K, BLK_M, IN_CH),
            z4, z4, z8, 0);
        __builtin_amdgcn_tensor_load_to_lds(
            tdm_group0(ldsB, gaB),
            tdm_group1(IN_CH, NPIX_PAD, BLK_K, BLK_N, IN_CH),
            z4, z4, z8, 0);
    };
    if (wid == 0) issue_pair(0, 0);            // prologue: prefetch step 0
#endif

    int phase = 0;
    for (int k0 = 0; k0 < IN_CH; k0 += BLK_K) {
        __syncthreads();   // all readers of buffer `phase^1` are done

#if USE_TDM
        if (wid == 0) {
            if (k0 + BLK_K < IN_CH) {
                issue_pair(k0 + BLK_K, phase ^ 1);   // prefetch next step
                // TENSORcnt completes in order: <=2 outstanding means the
                // current step's pair has landed; next step's DMA overlaps.
                __builtin_amdgcn_s_wait_tensorcnt(2);
            } else {
                __builtin_amdgcn_s_wait_tensorcnt(0);
            }
        }
        // TDM writes LDS behind the compiler's back: capture the LDS pointer
        // in a side-effecting asm so the fragment loads below cannot be
        // undef-folded or hoisted out of the loop.
        asm volatile("" : : "v"(&LDSBUF[0]) : "memory");
#else
        {   // ---- fallback staging: pure b128 traffic, no guards ----
            _Float16* Alds = LDSBUF + phase * BUF_HALVES;
            _Float16* Blds = Alds + A_LDS_HALVES;
            const int ar = tid >> 2;
            const int ac = (tid & 3) * 8;
            *(int4*)&Alds[ar * LDS_STRIDE + ac] =
                *(const int4*)(Ah + (size_t)(m0 + ar) * IN_CH + k0 + ac);
#pragma unroll
            for (int t = 0; t < 2; ++t) {
                const int idx = tid * 2 + t;
                const int br  = idx >> 2;
                const int bc  = (idx & 3) * 8;
                *(int4*)&Blds[br * LDS_STRIDE + bc] =
                    *(const int4*)(BhT + (size_t)(n0 + br) * IN_CH + k0 + bc);
            }
        }
#endif
        __syncthreads();

        const _Float16* Ab = LDSBUF + phase * BUF_HALVES;
        const _Float16* Bb = Ab + A_LDS_HALVES;

        // ---- A fragment (reused by all 4 WMMAs) ----
        union AF { v16h v; unsigned u[8]; } af;
#pragma unroll
        for (int i = 0; i < 4; ++i)
            af.u[i]     = *(const unsigned*)&Ab[lrow * LDS_STRIDE + a_klo + 2 * i];
#pragma unroll
        for (int i = 0; i < 4; ++i)
            af.u[4 + i] = *(const unsigned*)&Ab[lrow * LDS_STRIDE + a_khi + 2 * i];

#pragma unroll
        for (int f = 0; f < 4; ++f) {
            const int c = wave_n * 64 + f * 16 + (lane & 15);
            union BF { v16h v; unsigned u[8]; } bf;
#pragma unroll
            for (int i = 0; i < 8; ++i)
                bf.u[i] = *(const unsigned*)&Bb[c * LDS_STRIDE + b_klo + 2 * i + (i >= 4 ? 8 : 0)];
            acc[f] = __builtin_amdgcn_wmma_f32_16x16x32_f16(
                false, af.v, false, bf.v, (short)0, acc[f], false, false);
        }
        phase ^= 1;
    }

    // ---- epilogue: add bias, store fp32 score map ----
    const int mrow_base = m0 + wave_m * 16 + ((lane >> 4) << 3);
    float bv[8];
#pragma unroll
    for (int r = 0; r < 8; ++r) {
        const int m = mrow_base + r;
        bv[r] = (m < M_OUT) ? bias[m] : 0.f;
    }
#pragma unroll
    for (int f = 0; f < 4; ++f) {
        const int n = n0 + wave_n * 64 + f * 16 + (lane & 15);
        if (n < NPIX) {
#pragma unroll
            for (int r = 0; r < 8; ++r) {
                const int m = mrow_base + r;
                if (m < M_OUT) sm[(size_t)m * NPIX + n] = acc[f][r] + bv[r];
            }
        }
    }
}

// ---------------------------------------------------------------- PSROIAlign
__global__ __launch_bounds__(96)
void rfcn_psroi(const float* __restrict__ sm, const float* __restrict__ regions,
                float* __restrict__ out)
{
    const int r = blockIdx.x;
    const int t = threadIdx.x;
    if (t >= N_T) return;

    const float cy = regions[r * 4 + 0], cx = regions[r * 4 + 1];
    const float rh = regions[r * 4 + 2], rw = regions[r * 4 + 3];
    const float top  = (cy - rh * 0.5f) * (float)HFM;
    const float left = (cx - rw * 0.5f) * (float)WFM;
    const float bh = rh * (float)HFM / (float)KGRID;
    const float bw = rw * (float)WFM / (float)KGRID;

    float acc = 0.f;
    for (int ky = 0; ky < KGRID; ++ky) {
        for (int sy = 0; sy < 2; ++sy) {
            float y = top + ((float)ky + ((float)sy + 0.5f) * 0.5f) * bh;
            y = fminf(fmaxf(y, 0.f), (float)(HFM - 1));
            const float y0f = floorf(y);
            const int   y0  = (int)y0f;
            const int   y1  = min(y0 + 1, HFM - 1);
            const float wy1 = y - y0f, wy0 = 1.f - wy1;
            for (int kx = 0; kx < KGRID; ++kx) {
                const float* base = sm + (size_t)((t * KGRID + ky) * KGRID + kx) * NPIX;
                for (int sx = 0; sx < 2; ++sx) {
                    float x = left + ((float)kx + ((float)sx + 0.5f) * 0.5f) * bw;
                    x = fminf(fmaxf(x, 0.f), (float)(WFM - 1));
                    const float x0f = floorf(x);
                    const int   x0  = (int)x0f;
                    const int   x1  = min(x0 + 1, WFM - 1);
                    const float wx1 = x - x0f, wx0 = 1.f - wx1;
                    const float v00 = base[y0 * WFM + x0];
                    const float v01 = base[y0 * WFM + x1];
                    const float v10 = base[y1 * WFM + x0];
                    const float v11 = base[y1 * WFM + x1];
                    acc += wy0 * (wx0 * v00 + wx1 * v01)
                         + wy1 * (wx0 * v10 + wx1 * v11);
                }
            }
        }
    }
    out[r * N_T + t] = acc * (1.f / (float)(KGRID * KGRID * 2 * 2));
}

// ---------------------------------------------------------------- launch
extern "C" void kernel_launch(void* const* d_in, const int* in_sizes, int n_in,
                              void* d_out, int out_size, void* d_ws, size_t ws_size,
                              hipStream_t stream) {
    (void)in_sizes; (void)n_in; (void)out_size; (void)ws_size;
    const float* x  = (const float*)d_in[0];   // (1024,100,100)
    const float* rg = (const float*)d_in[1];   // (1024,4)
    const float* cw = (const float*)d_in[2];   // (3969,1024)
    const float* cb = (const float*)d_in[3];   // (3969,)

    // workspace layout: [Ah f16][BhT f16][sm f32]  (~188 MB)
    char* ws = (char*)d_ws;
    const size_t szA  = (size_t)M_PAD * IN_CH * sizeof(_Float16);
    const size_t offB = (szA + 255) & ~(size_t)255;
    const size_t szB  = (size_t)NPIX_PAD * IN_CH * sizeof(_Float16);
    const size_t offS = (offB + szB + 255) & ~(size_t)255;
    _Float16* Ah  = (_Float16*)(ws);
    _Float16* BhT = (_Float16*)(ws + offB);
    float*    sm  = (float*)(ws + offS);

    {   // fp32 -> f16 weights, zero-padded to M_PAD rows
        const int n = M_PAD * IN_CH;
        rfcn_cvt_w<<<(n + 255) / 256, 256, 0, stream>>>(cw, Ah, n);
    }
    {   // fp32 -> f16 transpose to pixel-major, zero-padded to NPIX_PAD rows
        dim3 grid(NPIX_PAD / 32, IN_CH / 32);
        rfcn_cvt_xT<<<grid, dim3(32, 8), 0, stream>>>(x, BhT);
    }
    {   // WMMA GEMM: grid (N tiles, M tiles) = (79, 63)
        dim3 grid(NPIX_PAD / BLK_N, M_PAD / BLK_M);
        rfcn_gemm_wmma<<<grid, 256, 0, stream>>>(Ah, BhT, cb, sm);
    }
    // PSROI pooling: one block per region
    rfcn_psroi<<<R_REG, 96, 0, stream>>>(sm, rg, (float*)d_out);
}
